// glob_feat_60490319396977
// MI455X (gfx1250) — matrix operand order repaired and checked by reading, not compiled, not run on hardware
//
#include <hip/hip_runtime.h>
#include <hip/hip_bf16.h>
#include <math.h>

// ---------------------------------------------------------------------------
// MI455X (gfx1250) implementation.
// Core math: V_WMMA_F32_16X16X32_BF16 (bf16 in, f32 accumulate).
// Weights double-buffered in LDS via GLOBAL_LOAD_ASYNC_TO_LDS_B128 (ASYNCcnt);
// B fragments software-pipelined (cur/next) to overlap ds_load with WMMA.
// ---------------------------------------------------------------------------

typedef __attribute__((ext_vector_type(16))) __bf16 v16bf;   // 32B WMMA A/B fragment per lane
typedef __attribute__((ext_vector_type(8)))  __bf16 bf16x8;  // 16B chunk
typedef __attribute__((ext_vector_type(8)))  float  v8f;     // WMMA C/D fragment per lane
typedef int v4i_ __attribute__((vector_size(16)));           // builtin's b128 unit

#define DIMX 128

#if defined(__HIP_DEVICE_COMPILE__) && \
    __has_builtin(__builtin_amdgcn_global_load_async_to_lds_b128) && \
    __has_builtin(__builtin_amdgcn_s_wait_asynccnt)
#define HAVE_ASYNC_LDS 1
#else
#define HAVE_ASYNC_LDS 0
#endif

// Global (AS1) / LDS (AS3) typed-pointer casts for the async builtin.
#define GPTR128(x) ((__attribute__((address_space(1))) v4i_*)(x))
#define LPTR128(x) ((__attribute__((address_space(3))) v4i_*)(x))

__device__ __forceinline__ v8f v8f_zero() {
    v8f z;
#pragma unroll
    for (int i = 0; i < 8; ++i) z[i] = 0.0f;
    return z;
}

__device__ __forceinline__ v16bf v16bf_zero() {
    v16bf z;
#pragma unroll
    for (int i = 0; i < 16; ++i) z[i] = (__bf16)0.0f;
    return z;
}

// Copy 32KB+32KB (W1[k], W2[k] fragments) global -> LDS, 256 threads.
// Async path: 16 GLOBAL_LOAD_ASYNC_TO_LDS_B128 per thread (coalesced 16B/lane).
__device__ __forceinline__ void stage_weights(const char* __restrict__ g1,
                                              const char* __restrict__ g2,
                                              char* lbuf, int tid) {
#pragma unroll
    for (int i = 0; i < 8; ++i) {
        const int off = (i * 256 + tid) * 16;
#if HAVE_ASYNC_LDS
        __builtin_amdgcn_global_load_async_to_lds_b128(
            GPTR128(g1 + off), LPTR128(lbuf + off), 0, 0);
        __builtin_amdgcn_global_load_async_to_lds_b128(
            GPTR128(g2 + off), LPTR128(lbuf + 32768 + off), 0, 0);
#else
        *(bf16x8*)(lbuf + off)         = *(const bf16x8*)(g1 + off);
        *(bf16x8*)(lbuf + 32768 + off) = *(const bf16x8*)(g2 + off);
#endif
    }
}

// ---------------------------------------------------------------------------
// Prep: feats f32 -> bf16.
// ---------------------------------------------------------------------------
__global__ __launch_bounds__(256)
void cvt_feats_kernel(const float* __restrict__ src, __bf16* __restrict__ dst, int total8) {
    int id = blockIdx.x * 256 + threadIdx.x;
    if (id >= total8) return;
    const float* s = src + (size_t)id * 8;
    bf16x8 p;
#pragma unroll
    for (int e = 0; e < 8; ++e) p[e] = (__bf16)s[e];
    *(bf16x8*)(dst + (size_t)id * 8) = p;
}

// ---------------------------------------------------------------------------
// Prep: swizzle [numK,128,128] f32 weights into WMMA B-fragment layout (bf16).
// Fragment (k, c=K-chunk, t=col-tile): lane L holds 16 bf16 =
//   B[K = c*32 + (e<8 ? h*8+e : 16+h*8+e-8), h=L>=16, N = t*16 + (L&15)]   (ISA 16-bit B 32x16).
// Fragment linear id == k*1024 + c*256 + t*32 + lane.
// ---------------------------------------------------------------------------
__global__ __launch_bounds__(256)
void swz_weights_kernel(const float* __restrict__ src, v16bf* __restrict__ dst, int total) {
    int id = blockIdx.x * 256 + threadIdx.x;
    if (id >= total) return;
    int lane = id & 31;
    int t    = (id >> 5) & 7;
    int c    = (id >> 8) & 3;
    int k    = id >> 10;
    int m    = lane & 15;
    int hi   = lane >> 4;
    const float* W = src + (size_t)k * DIMX * DIMX;
    int colN = t * 16 + m;
    v16bf v;
#pragma unroll
    for (int e = 0; e < 16; ++e) {
        int krow = c * 32 + (e < 8 ? hi * 8 + e : 16 + hi * 8 + (e - 8));
        v[e] = (__bf16)W[krow * DIMX + colN];
    }
    dst[id] = v;
}

__global__ __launch_bounds__(1024)
void zero_small_kernel(float* __restrict__ seg_sums, float* __restrict__ cnts) {
    int id = threadIdx.x;
    if (id < 1024) seg_sums[id] = 0.0f;
    if (id < 8)    cnts[id] = 0.0f;
}

__global__ __launch_bounds__(256)
void count_seg_kernel(const int* __restrict__ segid, float* __restrict__ cnts, int N) {
    __shared__ float c8[8];
    if (threadIdx.x < 8) c8[threadIdx.x] = 0.0f;
    __syncthreads();
    int i = blockIdx.x * 256 + threadIdx.x;
    if (i < N) atomicAdd(&c8[segid[i] & 7], 1.0f);
    __syncthreads();
    if (threadIdx.x < 8) atomicAdd(&cnts[threadIdx.x], c8[threadIdx.x]);
}

// ---------------------------------------------------------------------------
// Fused double sparse-conv.  256 threads = 8 waves; wave owns 16 nodes;
// block owns 128 nodes.  Per k: W1[k]+W2[k] staged in LDS (double buffer,
// async), A gathered as masked bf16 fragments, 64 WMMAs with pipelined B.
// Epilogue: bias+relu, LDS-staged coalesced bf16 stores, row means,
// sorted-segment partial sums.
// Dynamic LDS: 128KB (2 x 64KB weight buffers; first 64KB reused as f32 stage).
// ---------------------------------------------------------------------------
__global__ __launch_bounds__(256)
void conv2_wmma_kernel(const __bf16* __restrict__ xb,
                       const int* __restrict__ nidx,
                       const unsigned char* __restrict__ nmask,
                       const int* __restrict__ segid,
                       const v16bf* __restrict__ w1f,
                       const v16bf* __restrict__ w2f,
                       const float* __restrict__ bias1,
                       const float* __restrict__ bias2,
                       __bf16* __restrict__ out1,
                       __bf16* __restrict__ out2,
                       float* __restrict__ rowmean,
                       float* __restrict__ seg_sums,
                       int N, int K) {
    extern __shared__ char smraw[];          // 131072 bytes
    __shared__ int sseg[128];

    const int tid  = threadIdx.x;
    const int wave = tid >> 5;
    const int lane = tid & 31;
    const int base = blockIdx.x * 128;
    const int tbase = base + wave * 16;
    const int m  = lane & 15;                // A row (M) / C col (N)
    const int hi = lane >> 4;
    const int node = tbase + m;

    v8f acc1[8], acc2[8];
#pragma unroll
    for (int t = 0; t < 8; ++t) { acc1[t] = v8f_zero(); acc2[t] = v8f_zero(); }

    const char* w1b = (const char*)w1f;
    const char* w2b = (const char*)w2f;

    // Prologue: stage k=0 into buffer 0.
    stage_weights(w1b, w2b, smraw, tid);

    for (int k = 0; k < K; ++k) {
        char* lbuf = smraw + (size_t)(k & 1) * 65536;

        if (k + 1 < K) {
            char* nbuf = smraw + (size_t)((k + 1) & 1) * 65536;
            stage_weights(w1b + (size_t)(k + 1) * 32768,
                          w2b + (size_t)(k + 1) * 32768, nbuf, tid);
#if HAVE_ASYNC_LDS
            __builtin_amdgcn_s_wait_asynccnt(16);  // stage(k) done (in-order)
#endif
        } else {
#if HAVE_ASYNC_LDS
            __builtin_amdgcn_s_wait_asynccnt(0);
#endif
        }
        __syncthreads();                         // buffer k visible to all waves

        // ---- gather A fragments (16x32 bf16 per K-chunk), masked rows zeroed
        const int nbr = nidx[(size_t)node * K + k];
        const bool mk = nmask[(size_t)node * K + k] != 0;
        v16bf A[4];
        const __bf16* row = xb + (size_t)nbr * DIMX;
#pragma unroll
        for (int c = 0; c < 4; ++c) {
            if (mk) {
                bf16x8 lo  = *(const bf16x8*)(row + c * 32 + hi * 8);
                bf16x8 hi8 = *(const bf16x8*)(row + c * 32 + 16 + hi * 8);
                union { v16bf v; bf16x8 h[2]; } u;
                u.h[0] = lo; u.h[1] = hi8;
                A[c] = u.v;
            } else {
                A[c] = v16bf_zero();
            }
        }

        // ---- 8 col-tiles x 4 K-chunks x 2 convs WMMAs, pipelined B from LDS
        const v16bf* f1 = (const v16bf*)lbuf;
        const v16bf* f2 = (const v16bf*)(lbuf + 32768);
        v16bf b1c = f1[lane];                // idx 0: t=0,c=0 -> fi = lane
        v16bf b2c = f2[lane];
#pragma unroll
        for (int idx = 0; idx < 32; ++idx) {
            const int t = idx >> 2, c = idx & 3;
            v16bf b1n = b1c, b2n = b2c;
            if (idx + 1 < 32) {              // issue next loads before WMMAs
                const int t2 = (idx + 1) >> 2, c2 = (idx + 1) & 3;
                const int fin = c2 * 256 + t2 * 32 + lane;
                b1n = f1[fin];
                b2n = f2[fin];
            }
            acc1[t] = __builtin_amdgcn_wmma_f32_16x16x32_bf16(
                false, A[c], false, b1c, (short)0, acc1[t], false, false);
            acc2[t] = __builtin_amdgcn_wmma_f32_16x16x32_bf16(
                false, A[c], false, b2c, (short)0, acc2[t], false, false);
            b1c = b1n;
            b2c = b2n;
        }
        __syncthreads();   // all waves done reading buffer k before restaging
    }

    // ---------------- epilogue (reuse first 64KB of LDS as f32 stage) -------
    float* smem = (float*)smraw;

    // out1: bias + relu -> LDS
#pragma unroll
    for (int t = 0; t < 8; ++t) {
        const int col = t * 16 + m;
        const float b = bias1[col];
#pragma unroll
        for (int v = 0; v < 8; ++v) {
            const int r = wave * 16 + v + hi * 8;
            float val = acc1[t][v] + b;
            smem[r * DIMX + col] = val > 0.0f ? val : 0.0f;
        }
    }
    __syncthreads();

    {   // coalesced bf16 store of out1
        const int r = tid >> 1, h = tid & 1;
        const float* src = smem + r * DIMX + h * 64;
        __bf16* dst = out1 + (size_t)(base + r) * DIMX + h * 64;
#pragma unroll
        for (int j = 0; j < 64; j += 8) {
            bf16x8 p;
#pragma unroll
            for (int e = 0; e < 8; ++e) p[e] = (__bf16)src[j + e];
            *(bf16x8*)(dst + j) = p;
        }
    }
    if (tid < 128) {   // row means of out1 + stage segment ids
        const float* rsrc = smem + tid * DIMX;
        float s = 0.0f;
#pragma unroll 8
        for (int c = 0; c < DIMX; ++c) s += rsrc[c];
        rowmean[base + tid] = s * (1.0f / 128.0f);
        sseg[tid] = segid[base + tid] & 7;
    }
    __syncthreads();

    // out2: bias + relu -> LDS (reuse buffer)
#pragma unroll
    for (int t = 0; t < 8; ++t) {
        const int col = t * 16 + m;
        const float b = bias2[col];
#pragma unroll
        for (int v = 0; v < 8; ++v) {
            const int r = wave * 16 + v + hi * 8;
            float val = acc2[t][v] + b;
            smem[r * DIMX + col] = val > 0.0f ? val : 0.0f;
        }
    }
    __syncthreads();

    {   // coalesced bf16 store of out2
        const int r = tid >> 1, h = tid & 1;
        const float* src = smem + r * DIMX + h * 64;
        __bf16* dst = out2 + (size_t)(base + r) * DIMX + h * 64;
#pragma unroll
        for (int j = 0; j < 64; j += 8) {
            bf16x8 p;
#pragma unroll
            for (int e = 0; e < 8; ++e) p[e] = (__bf16)src[j + e];
            *(bf16x8*)(dst + j) = p;
        }
    }
    if (tid < 128) {   // segment partial sums of out2 (ids sorted)
        const int d = tid;
        int cur = sseg[0];
        float part = 0.0f;
        for (int r = 0; r < 128; ++r) {
            if (sseg[r] != cur) {
                atomicAdd(&seg_sums[cur * DIMX + d], part);
                part = 0.0f;
                cur = sseg[r];
            }
            part += smem[r * DIMX + d];
        }
        atomicAdd(&seg_sums[cur * DIMX + d], part);
    }
}

__global__ __launch_bounds__(256)
void seg_mean_kernel(const float* __restrict__ sums, const float* __restrict__ cnts,
                     float* __restrict__ mean) {
    int id = blockIdx.x * 256 + threadIdx.x;
    if (id < 1024) {
        float c = cnts[id >> 7];
        c = c < 1.0f ? 1.0f : c;
        mean[id] = sums[id] / c;
    }
}

// ---------------------------------------------------------------------------
// Epilogue kernel: mid = sqrt(rm*segmean + 1e-12) + out1 + out2 built directly
// into bf16 A fragments; mid @ W3 (32 WMMAs/wave, pipelined B);
// relu(feats - relu(.+b3)).
// ---------------------------------------------------------------------------
__global__ __launch_bounds__(256)
void final_wmma_kernel(const float* __restrict__ feats,
                       const __bf16* __restrict__ o1,
                       const __bf16* __restrict__ o2,
                       const float* __restrict__ rowmean,
                       const int* __restrict__ segid,
                       const float* __restrict__ segmean,
                       const v16bf* __restrict__ w3f,
                       const float* __restrict__ bias3,
                       float* __restrict__ out, int N) {
    __shared__ float smean[1024];
    for (int i = threadIdx.x; i < 1024; i += 256) smean[i] = segmean[i];
    __syncthreads();

    const int tid  = threadIdx.x;
    const int wave = tid >> 5;
    const int lane = tid & 31;
    const int tbase = blockIdx.x * 128 + wave * 16;
    const int m  = lane & 15;
    const int hi = lane >> 4;
    const int node = tbase + m;

    const float rm = rowmean[node];
    const int sg = segid[node] & 7;
    const float* smrow = smean + sg * DIMX;

    const __bf16* r1 = o1 + (size_t)node * DIMX;
    const __bf16* r2 = o2 + (size_t)node * DIMX;

    v16bf A[4];
#pragma unroll
    for (int c = 0; c < 4; ++c) {
        const int colLo = c * 32 + hi * 8;
        const int colHi = c * 32 + 16 + hi * 8;
        bf16x8 a1 = *(const bf16x8*)(r1 + colLo);
        bf16x8 b1 = *(const bf16x8*)(r1 + colHi);
        bf16x8 a2 = *(const bf16x8*)(r2 + colLo);
        bf16x8 b2 = *(const bf16x8*)(r2 + colHi);
        v16bf f;
#pragma unroll
        for (int e = 0; e < 8; ++e) {
            float enc = sqrtf(rm * smrow[colLo + e] + 1e-12f);
            f[e] = (__bf16)(enc + (float)a1[e] + (float)a2[e]);
        }
#pragma unroll
        for (int e = 0; e < 8; ++e) {
            float enc = sqrtf(rm * smrow[colHi + e] + 1e-12f);
            f[8 + e] = (__bf16)(enc + (float)b1[e] + (float)b2[e]);
        }
        A[c] = f;
    }

    v8f acc[8];
#pragma unroll
    for (int t = 0; t < 8; ++t) acc[t] = v8f_zero();

    v16bf bc = w3f[lane];                    // idx 0: t=0,c=0
#pragma unroll
    for (int idx = 0; idx < 32; ++idx) {
        const int t = idx >> 2, c = idx & 3;
        v16bf bn = bc;
        if (idx + 1 < 32) {
            const int t2 = (idx + 1) >> 2, c2 = (idx + 1) & 3;
            bn = w3f[c2 * 256 + t2 * 32 + lane];
        }
        acc[t] = __builtin_amdgcn_wmma_f32_16x16x32_bf16(
            false, A[c], false, bc, (short)0, acc[t], false, false);
        bc = bn;
    }

#pragma unroll
    for (int t = 0; t < 8; ++t) {
        const int col = t * 16 + m;
        const float b = bias3[col];
#pragma unroll
        for (int v = 0; v < 8; ++v) {
            const int n2 = tbase + v + hi * 8;
            float o3 = acc[t][v] + b;
            o3 = o3 > 0.0f ? o3 : 0.0f;
            float res = feats[(size_t)n2 * DIMX + col] - o3;
            out[(size_t)n2 * DIMX + col] = res > 0.0f ? res : 0.0f;
        }
    }
}

// ---------------------------------------------------------------------------
// Host launcher.  Inputs (setup_inputs order):
// 0 feats f32[N,128]  1 neighbor_idx i32[N,27]  2 neighbor_mask bool[N,27]
// 3 segment_ids i32[N]  4 num_segments(=8)  5 W1[27,128,128]  6 b1  7 W2
// 8 b2  9 W3[128,128]  10 b3.  Output f32[N,128].  Workspace ~310MB.
// ---------------------------------------------------------------------------
extern "C" void kernel_launch(void* const* d_in, const int* in_sizes, int n_in,
                              void* d_out, int out_size, void* d_ws, size_t ws_size,
                              hipStream_t stream) {
    const float*         feats = (const float*)d_in[0];
    const int*           nidx  = (const int*)d_in[1];
    const unsigned char* nmask = (const unsigned char*)d_in[2];   // jnp bool_ = 1 byte
    const int*           segid = (const int*)d_in[3];
    const float* W1 = (const float*)d_in[5];
    const float* b1 = (const float*)d_in[6];
    const float* W2 = (const float*)d_in[7];
    const float* b2 = (const float*)d_in[8];
    const float* W3 = (const float*)d_in[9];
    const float* b3 = (const float*)d_in[10];
    float* out = (float*)d_out;

    const int N = in_sizes[0] / DIMX;       // 400000 (divisible by 128)
    const int K = in_sizes[1] / N;          // 27

    char* ws = (char*)d_ws;
    auto a256 = [](size_t x) { return (x + 255) & ~(size_t)255; };
    size_t o = 0;
    __bf16* featsB = (__bf16*)(ws + o); o = a256(o + (size_t)N * DIMX * 2);
    v16bf* w1f = (v16bf*)(ws + o);      o = a256(o + (size_t)K * DIMX * DIMX * 2);
    v16bf* w2f = (v16bf*)(ws + o);      o = a256(o + (size_t)K * DIMX * DIMX * 2);
    v16bf* w3f = (v16bf*)(ws + o);      o = a256(o + (size_t)DIMX * DIMX * 2);
    __bf16* o1 = (__bf16*)(ws + o);     o = a256(o + (size_t)N * DIMX * 2);
    __bf16* o2 = (__bf16*)(ws + o);     o = a256(o + (size_t)N * DIMX * 2);
    float* rowmean  = (float*)(ws + o); o = a256(o + (size_t)N * 4);
    float* seg_sums = (float*)(ws + o); o = a256(o + 1024 * 4);
    float* cnts     = (float*)(ws + o); o = a256(o + 8 * 4);
    float* segmean  = (float*)(ws + o); o = a256(o + 1024 * 4);
    (void)ws_size; (void)n_in; (void)out_size;

    const int total8 = N * DIMX / 8;
    cvt_feats_kernel<<<(total8 + 255) / 256, 256, 0, stream>>>(feats, featsB, total8);
    swz_weights_kernel<<<(K * 1024 + 255) / 256, 256, 0, stream>>>(W1, w1f, K * 1024);
    swz_weights_kernel<<<(K * 1024 + 255) / 256, 256, 0, stream>>>(W2, w2f, K * 1024);
    swz_weights_kernel<<<4, 256, 0, stream>>>(W3, w3f, 1024);
    zero_small_kernel<<<1, 1024, 0, stream>>>(seg_sums, cnts);
    count_seg_kernel<<<(N + 255) / 256, 256, 0, stream>>>(segid, cnts, N);

    conv2_wmma_kernel<<<N / 128, 256, 131072, stream>>>(
        featsB, nidx, nmask, segid, w1f, w2f, b1, b2, o1, o2, rowmean, seg_sums, N, K);

    seg_mean_kernel<<<4, 256, 0, stream>>>(seg_sums, cnts, segmean);

    final_wmma_kernel<<<N / 128, 256, 0, stream>>>(
        feats, o1, o2, rowmean, segid, segmean, w3f, b3, out, N);
}
